// TransformerDecoder_13408887898353
// MI455X (gfx1250) — compile-verified, matrix operand order checked by reading, not run on hardware
//
#include <hip/hip_runtime.h>
#include <hip/hip_bf16.h>

// ---------------------------------------------------------------------------
// Types for CDNA5 WMMA (wave32, 16x16x32 bf16 -> f32) and TDM descriptors
// ---------------------------------------------------------------------------
typedef __attribute__((ext_vector_type(16))) __bf16 v16bf;
typedef __attribute__((ext_vector_type(8)))  __bf16 v8bf;
typedef __attribute__((ext_vector_type(8)))  float  v8f;
typedef __attribute__((ext_vector_type(4)))  float  v4f;
typedef __attribute__((ext_vector_type(4)))  unsigned int u32x4;
typedef __attribute__((ext_vector_type(8)))  int    i32x8;
typedef __attribute__((ext_vector_type(4)))  int    i32x4;

__device__ __forceinline__ __bf16 f2bf(float f) {
  unsigned u = __builtin_bit_cast(unsigned, f);
  u += 0x7FFFu + ((u >> 16) & 1u);          // round-to-nearest-even
  unsigned short s = (unsigned short)(u >> 16);
  return __builtin_bit_cast(__bf16, s);
}

// LDS byte offset of a __shared__ object (addrspace(3) pointers are 32-bit
// workgroup-relative, which is what the TDM D# lds_addr field wants).
__device__ __forceinline__ unsigned lds_off(const void* p) {
  return (unsigned)(size_t)(const __attribute__((address_space(3))) void*)p;
}

// Issue one TDM 2D tile load: rows x 32 f32 elements, row stride = K elems,
// from global tile base `g` into LDS at byte offset `lds`.
// D# layout per CDNA5 ISA ch.8 (group0: count/lds/global/type, group1: dims).
// This toolchain (clang-23 / therock-10.0) exposes the 6-arg builtin:
//   (uint32x4 g0, int32x8 g1, int32x4 g2, int32x4 g3, int32x8 extra, i32 cpol)
__device__ __forceinline__ void tdm_load_tile_f32(
    const float* g, unsigned lds, unsigned rows, unsigned tensor_cols,
    unsigned tensor_rows, unsigned stride_elems)
{
  unsigned long long ga = (unsigned long long)(size_t)g;
  u32x4 g0;
  g0[0] = 1u;                                             // count=1 (valid)
  g0[1] = lds;                                            // lds_addr (bytes)
  g0[2] = (unsigned)(ga & 0xFFFFFFFFu);                   // global_addr lo
  g0[3] = (unsigned)((ga >> 32) & 0x01FFFFFFu) | (2u << 30); // hi | type=2
  i32x8 g1;
  g1[0] = (int)(2u << 16);                                // data_size = 4B
  g1[1] = (int)((tensor_cols & 0xFFFFu) << 16);           // tensor_dim0 lo16
  g1[2] = (int)((tensor_cols >> 16) | ((tensor_rows & 0xFFFFu) << 16));
  g1[3] = (int)((tensor_rows >> 16) | (32u << 16));       // tile_dim0 = 32
  g1[4] = (int)rows;                                      // tile_dim1, tile_dim2=0
  g1[5] = (int)stride_elems;                              // dim0_stride lo32
  g1[6] = 0;                                              // stride hi, dim1_stride lo
  g1[7] = 0;
  i32x4 z4 = {};
  i32x8 z8 = {};
  __builtin_amdgcn_tensor_load_to_lds(g0, g1, z4, z4, z8, 0);
}

// ---------------------------------------------------------------------------
// Generic WMMA GEMM:  C[M,Nc] = act(A[M,K] @ W[Nc,K]^T + bias) (+ residual)
// Block tile 128x64, 8 waves (4x2), each wave 2x2 16x16 WMMA tiles, K step 32.
// Tiles are DMA'd into LDS (f32) by the Tensor Data Mover, converted to bf16
// while building fragments. M%128==0, Nc%64==0, K%32==0 by construction.
// ACT: 0=none, 1=gelu(tanh), 2=relu
// ---------------------------------------------------------------------------
template <int ACT, bool HASRES>
__global__ __launch_bounds__(256) void gemm_bf16_wmma(
    const float* __restrict__ A, const float* __restrict__ W,
    const float* __restrict__ bias, const float* __restrict__ residual,
    float* __restrict__ C, int M, int K, int Nc)
{
  __shared__ __align__(16) float Asf[128][32];   // 16 KB, TDM destination
  __shared__ __align__(16) float Bsf[64][32];    //  8 KB, TDM destination

  const int tid  = threadIdx.x;
  const int lane = tid & 31, wave = tid >> 5;
  const int wr = wave & 3, wc = wave >> 2;       // 4 row-waves x 2 col-waves
  const int ln = lane & 15, hh = lane >> 4;
  const int block_m = blockIdx.y * 128;
  const int block_n = blockIdx.x * 64;

  v8f acc[2][2] = {};

  for (int k0 = 0; k0 < K; k0 += 32) {
    // Tensor Data Mover: one DMA per tile, issued by wave 0 only.
    if (wave == 0) {
      tdm_load_tile_f32(A + (size_t)block_m * K + k0, lds_off(&Asf[0][0]),
                        128u, (unsigned)K, (unsigned)M, (unsigned)K);
      tdm_load_tile_f32(W + (size_t)block_n * K + k0, lds_off(&Bsf[0][0]),
                        64u, (unsigned)K, (unsigned)Nc, (unsigned)K);
      __builtin_amdgcn_s_wait_tensorcnt(0);
    }
    __syncthreads();

    // Fragment layout (ISA 7.12.2): lanes 0-15 hold K {0..7, 16..23},
    // lanes 16-31 hold K {8..15, 24..31}; same split for A and B frags.
    v16bf afr[2], bfr[2];
    #pragma unroll
    for (int i = 0; i < 2; ++i) {
      int m = wr * 32 + i * 16 + ln;
      v4f c0 = *(const v4f*)&Asf[m][8 * hh];
      v4f c1 = *(const v4f*)&Asf[m][8 * hh + 4];
      v4f c2 = *(const v4f*)&Asf[m][16 + 8 * hh];
      v4f c3 = *(const v4f*)&Asf[m][16 + 8 * hh + 4];
      #pragma unroll
      for (int e = 0; e < 4; ++e) {
        afr[i][e]      = f2bf(c0[e]);
        afr[i][e + 4]  = f2bf(c1[e]);
        afr[i][e + 8]  = f2bf(c2[e]);
        afr[i][e + 12] = f2bf(c3[e]);
      }
    }
    #pragma unroll
    for (int j = 0; j < 2; ++j) {
      int n = wc * 32 + j * 16 + ln;
      v4f c0 = *(const v4f*)&Bsf[n][8 * hh];
      v4f c1 = *(const v4f*)&Bsf[n][8 * hh + 4];
      v4f c2 = *(const v4f*)&Bsf[n][16 + 8 * hh];
      v4f c3 = *(const v4f*)&Bsf[n][16 + 8 * hh + 4];
      #pragma unroll
      for (int e = 0; e < 4; ++e) {
        bfr[j][e]      = f2bf(c0[e]);
        bfr[j][e + 4]  = f2bf(c1[e]);
        bfr[j][e + 8]  = f2bf(c2[e]);
        bfr[j][e + 12] = f2bf(c3[e]);
      }
    }
    #pragma unroll
    for (int i = 0; i < 2; ++i)
      #pragma unroll
      for (int j = 0; j < 2; ++j)
        acc[i][j] = __builtin_amdgcn_wmma_f32_16x16x32_bf16(
            false, afr[i], false, bfr[j], (short)0, acc[i][j], false, false);
    __syncthreads();
  }

  // Straight-line epilogue (ACT/HASRES compile-time).
  #pragma unroll
  for (int i = 0; i < 2; ++i)
    #pragma unroll
    for (int j = 0; j < 2; ++j)
      #pragma unroll
      for (int r = 0; r < 8; ++r) {
        int m = block_m + wr * 32 + i * 16 + r + 8 * hh;
        int n = block_n + wc * 32 + j * 16 + ln;
        float v = acc[i][j][r] + bias[n];
        if constexpr (ACT == 1) {
          float inner = 0.7978845608f * (v + 0.044715f * v * v * v);
          v = 0.5f * v * (1.0f + tanhf(inner));
        } else if constexpr (ACT == 2) {
          v = fmaxf(v, 0.0f);
        }
        if constexpr (HASRES) v += residual[(size_t)m * Nc + n];
        C[(size_t)m * Nc + n] = v;
      }
}

// ---------------------------------------------------------------------------
// Flash self-attention. qkv: [N,768] (q|k|v, each 8 heads x 32 dims).
// Grid: B*NH*(Q/64)=1024 blocks of 128 threads; each wave owns 16 queries.
// Computes S^T = K*Q^T so softmax is lane-local; exp(S^T) of two key tiles is
// directly the 32x16 B-fragment of the O^T = V^T * P^T WMMA.
// ---------------------------------------------------------------------------
__global__ __launch_bounds__(128) void self_attn_flash(
    const float* __restrict__ qkv, float* __restrict__ out)
{
  __shared__ __align__(16) __bf16 Ks[32][32];   // (key, dim) bf16
  __shared__ __align__(16) __bf16 Vt[32][32];   // (dim, key) bf16

  const int tid = threadIdx.x, lane = tid & 31, wave = tid >> 5;
  const int blk = blockIdx.x;
  const int qb = blk & 15, bh = blk >> 4;
  const int b = bh >> 3, h = bh & 7;
  const int ln = lane & 15, hh = lane >> 4;
  const int qrow = b * 1024 + qb * 64 + wave * 16 + ln;
  const float scale = 0.17677669529663687f;     // 1/sqrt(32)

  // Q^T B-fragment (query = lane column, dims along K)
  const float* qp = qkv + (size_t)qrow * 768 + h * 32;
  v16bf bq;
  #pragma unroll
  for (int e = 0; e < 8; ++e) {
    bq[e]     = f2bf(qp[8 * hh + e] * scale);
    bq[e + 8] = f2bf(qp[16 + 8 * hh + e] * scale);
  }

  v8f o0 = {}, o1 = {};                         // O^T dim tiles 0-15 / 16-31
  float mrun = -1e30f, srun = 0.0f;

  for (int kb = 0; kb < 1024; kb += 32) {
    // cooperative stage of 32 keys (K row-major, V transposed) as bf16
    #pragma unroll
    for (int i = 0; i < 8; ++i) {
      int idx = tid * 8 + i;
      int key = idx >> 5, dim = idx & 31;
      const float* row = qkv + (size_t)(b * 1024 + kb + key) * 768 + h * 32;
      Ks[key][dim] = f2bf(row[256 + dim]);
      Vt[dim][key] = f2bf(row[512 + dim]);
    }
    __syncthreads();

    // A-frags: K tiles (16 keys x 32 dims), rows = keys
    v16bf a0, a1;
    {
      v8bf l0 = *(const v8bf*)&Ks[ln][8 * hh];
      v8bf h0 = *(const v8bf*)&Ks[ln][16 + 8 * hh];
      v8bf l1 = *(const v8bf*)&Ks[ln + 16][8 * hh];
      v8bf h1 = *(const v8bf*)&Ks[ln + 16][16 + 8 * hh];
      #pragma unroll
      for (int e = 0; e < 8; ++e) {
        a0[e] = l0[e]; a0[e + 8] = h0[e];
        a1[e] = l1[e]; a1[e + 8] = h1[e];
      }
    }
    v8f z = {};
    v8f s0 = __builtin_amdgcn_wmma_f32_16x16x32_bf16(false, a0, false, bq, (short)0, z, false, false);
    v8f s1 = __builtin_amdgcn_wmma_f32_16x16x32_bf16(false, a1, false, bq, (short)0, z, false, false);

    // online softmax (scores for this lane's 16-key subset; merge halves)
    float lm = -1e30f;
    #pragma unroll
    for (int r = 0; r < 8; ++r) lm = fmaxf(lm, fmaxf(s0[r], s1[r]));
    lm = fmaxf(lm, __shfl_xor(lm, 16, 32));
    float nm = fmaxf(mrun, lm);
    float fac = __expf(mrun - nm);
    #pragma unroll
    for (int r = 0; r < 8; ++r) { o0[r] *= fac; o1[r] *= fac; }
    srun *= fac;

    // exp(S^T) packs directly into the 32x16 B-fragment for P*V
    v16bf pB;
    #pragma unroll
    for (int e = 0; e < 8; ++e) {
      float p0 = __expf(s0[e] - nm);
      float p1 = __expf(s1[e] - nm);
      srun += p0 + p1;
      pB[e] = f2bf(p0); pB[e + 8] = f2bf(p1);
    }
    mrun = nm;

    // A-frags: V^T dim tiles (16 dims x 32 keys)
    v16bf va0, va1;
    {
      v8bf l0 = *(const v8bf*)&Vt[ln][8 * hh];
      v8bf h0 = *(const v8bf*)&Vt[ln][16 + 8 * hh];
      v8bf l1 = *(const v8bf*)&Vt[ln + 16][8 * hh];
      v8bf h1 = *(const v8bf*)&Vt[ln + 16][16 + 8 * hh];
      #pragma unroll
      for (int e = 0; e < 8; ++e) {
        va0[e] = l0[e]; va0[e + 8] = h0[e];
        va1[e] = l1[e]; va1[e + 8] = h1[e];
      }
    }
    o0 = __builtin_amdgcn_wmma_f32_16x16x32_bf16(false, va0, false, pB, (short)0, o0, false, false);
    o1 = __builtin_amdgcn_wmma_f32_16x16x32_bf16(false, va1, false, pB, (short)0, o1, false, false);
    __syncthreads();
  }

  float stot = srun + __shfl_xor(srun, 16, 32);
  float inv = 1.0f / stot;
  float* orow = out + (size_t)qrow * 256 + h * 32;
  #pragma unroll
  for (int r = 0; r < 8; ++r) {
    orow[r + 8 * hh]      = o0[r] * inv;
    orow[16 + r + 8 * hh] = o1[r] * inv;
  }
}

// ---------------------------------------------------------------------------
// Positional encodings: x += sincos(pix/192) + sincos(subpix)
// ---------------------------------------------------------------------------
__global__ void posenc_add(float* __restrict__ x, const int* __restrict__ idxb,
                           const float* __restrict__ subb)
{
  int gid = blockIdx.x * 256 + threadIdx.x;
  int t = gid >> 8, c = gid & 255;
  int a = c >> 7, cc = c & 127, i = cc & 63, isc = cc >> 6;
  float freq = __expf(-(float)i * (9.2103403719762f / 64.0f)); // 10000^{-i/64}
  float cp = (float)idxb[t * 3 + 1 + a] * (1.0f / 192.0f);
  float cs = subb[t * 2 + a];
  float a1 = cp * freq, a2 = cs * freq;
  float e1 = isc ? __cosf(a1) : __sinf(a1);
  float e2 = isc ? __cosf(a2) : __sinf(a2);
  x[gid] += e1 + e2;
}

// ---------------------------------------------------------------------------
// LayerNorm over 256 channels; one wave per row, 8 rows per block.
// ---------------------------------------------------------------------------
__global__ __launch_bounds__(256) void layernorm256(
    const float* __restrict__ in, float* __restrict__ out,
    const float* __restrict__ g, const float* __restrict__ bta)
{
  int row = blockIdx.x * 8 + (threadIdx.x >> 5);
  int lane = threadIdx.x & 31;
  const float* r = in + (size_t)row * 256;
  float v[8]; float s = 0.0f;
  #pragma unroll
  for (int i = 0; i < 8; ++i) { v[i] = r[lane + 32 * i]; s += v[i]; }
  #pragma unroll
  for (int off = 16; off; off >>= 1) s += __shfl_xor(s, off, 32);
  float mean = s * (1.0f / 256.0f);
  float var = 0.0f;
  #pragma unroll
  for (int i = 0; i < 8; ++i) { float d = v[i] - mean; var += d * d; }
  #pragma unroll
  for (int off = 16; off; off >>= 1) var += __shfl_xor(var, off, 32);
  float rs = rsqrtf(var * (1.0f / 256.0f) + 1e-5f);
  float* w = out + (size_t)row * 256;
  #pragma unroll
  for (int i = 0; i < 8; ++i) {
    int c = lane + 32 * i;
    w[c] = (v[i] - mean) * rs * g[c] + bta[c];
  }
}

// ---------------------------------------------------------------------------
// Cross-attn key gather + bilinear rel-pos table. One block per (token,offset)
// within a chunk, 256 threads = channels. Writes keys and keys+rpe.
// ---------------------------------------------------------------------------
__global__ __launch_bounds__(256) void gather_rpe(
    const float* __restrict__ image, const int* __restrict__ idxb,
    const float* __restrict__ subb, const float* __restrict__ table,
    float* __restrict__ keys, float* __restrict__ keysrpe, int tok0)
{
  int blk = blockIdx.x;                 // CH*25
  int t = tok0 + blk / 25;
  int o = blk % 25;
  int d = threadIdx.x;
  int oy = o / 5 - 2, ox = o % 5 - 2;
  int b = idxb[t * 3 + 0];
  int y = idxb[t * 3 + 1] + oy; y = y < 0 ? 0 : (y > 191 ? 191 : y);
  int x = idxb[t * 3 + 2] + ox; x = x < 0 ? 0 : (x > 191 ? 191 : x);
  float iy = (float)oy - subb[t * 2 + 0] + 3.0f;   // rel + (P//2 + 1)
  float ix = (float)ox - subb[t * 2 + 1] + 3.0f;
  int i0y = (int)fminf(fmaxf(floorf(iy), 0.0f), 5.0f);
  int i0x = (int)fminf(fmaxf(floorf(ix), 0.0f), 5.0f);
  float fy = iy - (float)i0y, fx = ix - (float)i0x;
  int i1y = i0y + 1 > 5 ? 5 : i0y + 1;
  int i1x = i0x + 1 > 5 ? 5 : i0x + 1;
  float t00 = table[(i0y * 6 + i0x) * 256 + d];
  float t01 = table[(i0y * 6 + i1x) * 256 + d];
  float t10 = table[(i1y * 6 + i0x) * 256 + d];
  float t11 = table[(i1y * 6 + i1x) * 256 + d];
  float rpe = (1 - fy) * (1 - fx) * t00 + (1 - fy) * fx * t01 +
              fy * (1 - fx) * t10 + fy * fx * t11;
  float kv = image[(((size_t)b * 192 + y) * 192 + x) * 256 + d];
  size_t row = (size_t)blk * 256;
  keys[row + d] = kv;
  keysrpe[row + d] = kv + rpe;
}

// ---------------------------------------------------------------------------
// Cross-attention core: 1 query x 25 keys x 32 dims per (token, head) thread.
// ---------------------------------------------------------------------------
__global__ void cross_attn_small(
    const float* __restrict__ qca, const float* __restrict__ kp,
    const float* __restrict__ vp, float* __restrict__ out, int tok0)
{
  int gid = blockIdx.x * blockDim.x + threadIdx.x;   // CH*8
  int tl = gid >> 3, h = gid & 7;
  int t = tok0 + tl;
  const float* q = qca + (size_t)t * 256 + h * 32;
  float qr[32];
  #pragma unroll
  for (int d = 0; d < 32; ++d) qr[d] = q[d];
  float sc[25], m = -1e30f;
  for (int j = 0; j < 25; ++j) {
    const float* kr = kp + ((size_t)tl * 25 + j) * 256 + h * 32;
    float s = 0.0f;
    #pragma unroll
    for (int d = 0; d < 32; ++d) s += qr[d] * kr[d];
    s *= 0.17677669529663687f;
    sc[j] = s; m = fmaxf(m, s);
  }
  float ssum = 0.0f;
  for (int j = 0; j < 25; ++j) { sc[j] = __expf(sc[j] - m); ssum += sc[j]; }
  float inv = 1.0f / ssum;
  float o[32];
  #pragma unroll
  for (int d = 0; d < 32; ++d) o[d] = 0.0f;
  for (int j = 0; j < 25; ++j) {
    const float* vr = vp + ((size_t)tl * 25 + j) * 256 + h * 32;
    float p = sc[j] * inv;
    #pragma unroll
    for (int d = 0; d < 32; ++d) o[d] += p * vr[d];
  }
  float* orow = out + (size_t)t * 256 + h * 32;
  #pragma unroll
  for (int d = 0; d < 32; ++d) orow[d] = o[d];
}

// ---------------------------------------------------------------------------
// Position-delta head tail + index/subpixel update.
// ---------------------------------------------------------------------------
__global__ void pos_update(const float* __restrict__ hrelu,
                           const float* __restrict__ w2, const float* __restrict__ b2,
                           int* __restrict__ idxb, float* __restrict__ subb)
{
  int t = blockIdx.x * blockDim.x + threadIdx.x;
  if (t >= 8192) return;
  const float* hr = hrelu + (size_t)t * 256;
  float d0 = b2[0], d1 = b2[1];
  for (int k = 0; k < 256; ++k) {
    float hv = hr[k];
    d0 += hv * w2[k];
    d1 += hv * w2[256 + k];
  }
  #pragma unroll
  for (int a = 0; a < 2; ++a) {
    float delta = a ? d1 : d0;
    float np = (float)idxb[t * 3 + 1 + a] + subb[t * 2 + a] + delta;
    float fl = floorf(np);
    idxb[t * 3 + 1 + a] = (int)fl;
    subb[t * 2 + a] = np - fl;
  }
}

// ---------------------------------------------------------------------------
// Pack tuple output: (x[N,256] f32, indices[N,3] i32, subpix[N,2] f32)
// ---------------------------------------------------------------------------
__global__ void finalize_out(const float* __restrict__ x, const int* __restrict__ idxb,
                             const float* __restrict__ subb, float* __restrict__ outf)
{
  int gid = blockIdx.x * 256 + threadIdx.x;
  const int NX = 8192 * 256;
  if (gid < NX) outf[gid] = x[gid];
  if (gid < 8192 * 3) ((int*)outf)[NX + gid] = idxb[gid];
  if (gid < 8192 * 2) outf[NX + 8192 * 3 + gid] = subb[gid];
}

// ---------------------------------------------------------------------------
// Host orchestration
// ---------------------------------------------------------------------------
extern "C" void kernel_launch(void* const* d_in, const int* in_sizes, int n_in,
                              void* d_out, int out_size, void* d_ws, size_t ws_size,
                              hipStream_t stream) {
  (void)in_sizes; (void)n_in; (void)out_size; (void)ws_size;
  const int N = 8192, D = 256, FF = 1024;
  const size_t ND = (size_t)N * D;
  const int CH = 512, CHROWS = CH * 25;       // cross-attn token chunk
  const size_t CHK = (size_t)CHROWS * 256;

  float* x    = (float*)d_ws;
  float* xn   = x + ND;
  float* qkv  = xn + ND;                       // also holds q_ca (first N*256)
  float* attn = qkv + (size_t)N * 768;
  float* hff  = attn + ND;
  float* keys    = hff + (size_t)N * FF;
  float* keysrpe = keys + CHK;
  float* kproj   = keysrpe + CHK;
  float* vproj   = kproj + CHK;
  int*   idxb = (int*)(vproj + CHK);
  float* subb = (float*)(idxb + (size_t)N * 3);

  const float* image = (const float*)d_in[4];

  (void)hipMemcpyAsync(x, d_in[0], ND * sizeof(float), hipMemcpyDeviceToDevice, stream);
  (void)hipMemcpyAsync(idxb, d_in[1], (size_t)N * 3 * sizeof(int), hipMemcpyDeviceToDevice, stream);
  (void)hipMemcpyAsync(subb, d_in[2], (size_t)N * 2 * sizeof(float), hipMemcpyDeviceToDevice, stream);

  auto gemm = [&](const float* A, const float* W, const float* bias,
                  const float* res, float* C, int M, int K, int Nc, int act) {
    dim3 grid(Nc / 64, M / 128);
    if (res) {
      gemm_bf16_wmma<0, true><<<grid, 256, 0, stream>>>(A, W, bias, res, C, M, K, Nc);
    } else if (act == 1) {
      gemm_bf16_wmma<1, false><<<grid, 256, 0, stream>>>(A, W, bias, nullptr, C, M, K, Nc);
    } else if (act == 2) {
      gemm_bf16_wmma<2, false><<<grid, 256, 0, stream>>>(A, W, bias, nullptr, C, M, K, Nc);
    } else {
      gemm_bf16_wmma<0, false><<<grid, 256, 0, stream>>>(A, W, bias, nullptr, C, M, K, Nc);
    }
  };

  for (int l = 0; l < 4; ++l) {
    const int base = 5 + 23 * l;
    const float* sa_lng = (const float*)d_in[base + 0];
    const float* sa_lnb = (const float*)d_in[base + 1];
    const float* in_w   = (const float*)d_in[base + 2];
    const float* in_b   = (const float*)d_in[base + 3];
    const float* sa_ow  = (const float*)d_in[base + 4];
    const float* sa_ob  = (const float*)d_in[base + 5];
    const float* q_w    = (const float*)d_in[base + 8];
    const float* q_b    = (const float*)d_in[base + 9];
    const float* k_w    = (const float*)d_in[base + 10];
    const float* k_b    = (const float*)d_in[base + 11];
    const float* v_w    = (const float*)d_in[base + 12];
    const float* v_b    = (const float*)d_in[base + 13];
    const float* ca_ow  = (const float*)d_in[base + 14];
    const float* ca_ob  = (const float*)d_in[base + 15];
    const float* rtab   = (const float*)d_in[base + 16];
    const float* f_lng  = (const float*)d_in[base + 17];
    const float* f_lnb  = (const float*)d_in[base + 18];
    const float* w1     = (const float*)d_in[base + 19];
    const float* b1     = (const float*)d_in[base + 20];
    const float* w2     = (const float*)d_in[base + 21];
    const float* b2     = (const float*)d_in[base + 22];

    // x = queries + sincos(pix) + sincos(sub)
    posenc_add<<<N, 256, 0, stream>>>(x, idxb, subb);

    // ---- self-attention ----
    layernorm256<<<N / 8, 256, 0, stream>>>(x, xn, sa_lng, sa_lnb);
    gemm(xn, in_w, in_b, nullptr, qkv, N, 256, 768, 0);
    self_attn_flash<<<1024, 128, 0, stream>>>(qkv, attn);
    gemm(attn, sa_ow, sa_ob, x, x, N, 256, 256, 0);

    // ---- cross-attention ----
    gemm(x, q_w, q_b, nullptr, qkv /* q_ca */, N, 256, 256, 0);
    for (int c = 0; c < N / CH; ++c) {
      int tok0 = c * CH;
      gather_rpe<<<CHROWS, 256, 0, stream>>>(image, idxb, subb, rtab, keys, keysrpe, tok0);
      gemm(keysrpe, k_w, k_b, nullptr, kproj, CHROWS, 256, 256, 0);
      gemm(keys,    v_w, v_b, nullptr, vproj, CHROWS, 256, 256, 0);
      cross_attn_small<<<(CH * 8) / 256, 256, 0, stream>>>(qkv, kproj, vproj, attn, tok0);
    }
    gemm(attn, ca_ow, ca_ob, x, x, N, 256, 256, 0);

    // ---- FFN ----
    layernorm256<<<N / 8, 256, 0, stream>>>(x, xn, f_lng, f_lnb);
    gemm(xn, w1, b1, nullptr, hff, N, 256, FF, 1 /* gelu */);
    gemm(hff, w2, b2, x, x, N, FF, 256, 0);

    // ---- position head + state update ----
    const float* dp_w1 = (const float*)d_in[97];
    const float* dp_b1 = (const float*)d_in[98];
    const float* dp_w2 = (const float*)d_in[99];
    const float* dp_b2 = (const float*)d_in[100];
    gemm(x, dp_w1, dp_b1, nullptr, xn, N, 256, 256, 2 /* relu */);
    pos_update<<<N / 256, 256, 0, stream>>>(xn, dp_w2, dp_b2, idxb, subb);
  }

  finalize_out<<<N, 256, 0, stream>>>(x, idxb, subb, (float*)d_out);
}
